// NormEMAVectorQuantizer_29463475650758
// MI455X (gfx1250) — matrix operand order repaired and checked by reading, not compiled, not run on hardware
//
#include <hip/hip_runtime.h>
#include <math.h>

// ---------------------------------------------------------------------------
// NormEMAVectorQuantizer eval forward for MI455X (gfx1250, wave32, WMMA bf16)
//   z:      [32768, 256] f32
//   weight: [8192, 256]  f32 (l2-normalized)
// Outputs (concatenated in d_out, f32):
//   z_q_st [32768*256], loss [1], encoding_indices [32768] (as float values)
// ---------------------------------------------------------------------------

#define BB 32768
#define DD 256
#define NN 8192
#define BETA_C 1.0f
#define EPS_C 1e-12f

typedef __bf16 bf16_t;
typedef __attribute__((ext_vector_type(16))) __bf16 v16bf;
typedef __attribute__((ext_vector_type(8)))  __bf16 v8bf;
typedef __attribute__((ext_vector_type(8)))  float  v8f;

union Frag16 { v16bf v; v8bf h[2]; };

// Workspace layout (bytes), all 256B aligned.
#define OFF_ZB     ((size_t)0)                         // bf16 -2*z_norm : 16 MB
#define OFF_WB     ((size_t)(OFF_ZB   + (size_t)BB*DD*2)) // bf16 weight : 4 MB
#define OFF_WN2    ((size_t)(OFF_WB   + (size_t)NN*DD*2)) // f32 ||w||^2 : 32 KB
#define OFF_ZNORM  ((size_t)(OFF_WN2  + (size_t)NN*4))    // f32 z_norm  : 32 MB
#define OFF_IDX    ((size_t)(OFF_ZNORM+ (size_t)BB*DD*4)) // int idx     : 128 KB
#define OFF_PART   ((size_t)(OFF_IDX  + (size_t)BB*4))    // f32 partial : 128 KB

// ---------------- block reduction helper (256 threads, wave32) -------------
__device__ __forceinline__ float block_reduce_256(float v, float* red) {
  const int t = threadIdx.x;
#pragma unroll
  for (int off = 16; off > 0; off >>= 1) v += __shfl_xor(v, off, 32);
  if ((t & 31) == 0) red[t >> 5] = v;
  __syncthreads();
  if (t == 0) {
    float tot = 0.f;
#pragma unroll
    for (int i = 0; i < 8; ++i) tot += red[i];
    red[0] = tot;
  }
  __syncthreads();
  return red[0];
}

// ---------------- prep: normalize z, emit f32 z_norm + bf16(-2*z_norm) ------
__global__ __launch_bounds__(256) void vq_prep_z(const float* __restrict__ z,
                                                 float* __restrict__ znorm,
                                                 bf16_t* __restrict__ zb) {
  __shared__ float red[8];
  const size_t row = blockIdx.x;
  const int t = threadIdx.x;
  const float x = z[row * DD + t];
  const float ss = block_reduce_256(x * x, red);
  const float inv = 1.0f / sqrtf(ss + EPS_C);
  const float zn = x * inv;
  znorm[row * DD + t] = zn;
  zb[row * DD + t] = (bf16_t)(-2.0f * zn);
}

// ---------------- prep: bf16 weight + exact f32 ||w||^2 ---------------------
__global__ __launch_bounds__(256) void vq_prep_w(const float* __restrict__ w,
                                                 float* __restrict__ wn2,
                                                 bf16_t* __restrict__ wb) {
  __shared__ float red[8];
  const size_t row = blockIdx.x;
  const int t = threadIdx.x;
  const float x = w[row * DD + t];
  const float ss = block_reduce_256(x * x, red);
  wb[row * DD + t] = (bf16_t)x;
  if (t == 0) wn2[row] = ss;
}

// ---------------- core: WMMA bf16 GEMM fused with row argmin ----------------
// Block = 64 threads (2 waves); each wave owns 32 rows (2 strips of 16),
// A (K=256) register resident, streams all N=8192 codes in 16-col tiles.
// Score(m,n) = ||w_n||^2 + dot(-2*z_m, w_n); C initialized with ||w_n||^2.
__global__ __launch_bounds__(64) void vq_gemm_argmin(
    const bf16_t* __restrict__ zb, const bf16_t* __restrict__ wb,
    const float* __restrict__ wn2, int* __restrict__ outIdx) {
  const int lane   = threadIdx.x & 31;
  const int wave   = threadIdx.x >> 5;
  const int laneLo = lane & 15;
  const int hi     = lane >> 4;   // half-wave select
  const int aOff   = hi * 8;      // A frag: lanes>=16 hold K+8 / K+24
  const int bOff   = hi * 8;      // B frag: lanes>=16 hold K+8 / K+24

  const int rowBase = blockIdx.x * 64 + wave * 32;

  // Preload A fragments: 2 strips x 8 k-chunks (16x32 bf16 each).
  v16bf a[2][8];
#pragma unroll
  for (int s = 0; s < 2; ++s) {
    const bf16_t* ap = zb + (size_t)(rowBase + s * 16 + laneLo) * DD;
#pragma unroll
    for (int k = 0; k < 8; ++k) {
      Frag16 f;
      f.h[0] = *(const v8bf*)(ap + k * 32 + aOff);
      f.h[1] = *(const v8bf*)(ap + k * 32 + 16 + aOff);
      a[s][k] = f.v;
    }
  }

  float minv[2][8];
  int   mini[2][8];
#pragma unroll
  for (int s = 0; s < 2; ++s)
#pragma unroll
    for (int v = 0; v < 8; ++v) { minv[s][v] = 3.0e38f; mini[s][v] = 0; }

  for (int nb = 0; nb < NN; nb += 16) {
    const float w2 = wn2[nb + laneLo];
    const bf16_t* bp = wb + (size_t)(nb + laneLo) * DD;

    v8f c00, c01, c10, c11;  // 4 independent accumulation chains
#pragma unroll
    for (int i = 0; i < 8; ++i) {
      c00[i] = w2; c10[i] = w2; c01[i] = 0.f; c11[i] = 0.f;
    }

#pragma unroll
    for (int k = 0; k < 8; ++k) {
      Frag16 f;
      f.h[0] = *(const v8bf*)(bp + k * 32 + bOff);
      f.h[1] = *(const v8bf*)(bp + k * 32 + 16 + bOff);
      const v16bf bk = f.v;
      if (k < 4) {
        c00 = __builtin_amdgcn_wmma_f32_16x16x32_bf16(false, a[0][k], false, bk,
                                                      (short)0, c00, false, false);
        c10 = __builtin_amdgcn_wmma_f32_16x16x32_bf16(false, a[1][k], false, bk,
                                                      (short)0, c10, false, false);
      } else {
        c01 = __builtin_amdgcn_wmma_f32_16x16x32_bf16(false, a[0][k], false, bk,
                                                      (short)0, c01, false, false);
        c11 = __builtin_amdgcn_wmma_f32_16x16x32_bf16(false, a[1][k], false, bk,
                                                      (short)0, c11, false, false);
      }
    }

    const int n = nb + laneLo;
#pragma unroll
    for (int v = 0; v < 8; ++v) {
      const float s0 = c00[v] + c01[v];
      if (s0 < minv[0][v]) { minv[0][v] = s0; mini[0][v] = n; }
      const float s1 = c10[v] + c11[v];
      if (s1 < minv[1][v]) { minv[1][v] = s1; mini[1][v] = n; }
    }
  }

  // Reduce argmin across the 16 column-slot lanes (xor 1..8 keeps half-waves).
#pragma unroll
  for (int s = 0; s < 2; ++s)
#pragma unroll
    for (int v = 0; v < 8; ++v) {
      float mv = minv[s][v];
      int   mi = mini[s][v];
#pragma unroll
      for (int off = 1; off < 16; off <<= 1) {
        const float ov = __shfl_xor(mv, off, 32);
        const int   oi = __shfl_xor(mi, off, 32);
        if (ov < mv || (ov == mv && oi < mi)) { mv = ov; mi = oi; }
      }
      if (laneLo == 0) outIdx[rowBase + s * 16 + hi * 8 + v] = mi;
    }
}

// ---------------- gather z_q, exact f32 loss partials, indices --------------
__global__ __launch_bounds__(256) void vq_gather(
    const float* __restrict__ weight, const float* __restrict__ znorm,
    const int* __restrict__ idx, float* __restrict__ outZq,
    float* __restrict__ outIdxF, float* __restrict__ partial) {
  __shared__ float red[8];
  const size_t row = blockIdx.x;
  const int t = threadIdx.x;
  const int id = idx[row];
  const float w  = weight[(size_t)id * DD + t];
  const float zn = znorm[row * DD + t];
  const float d  = w - zn;
  outZq[row * DD + t] = zn + d;  // matches z + (z_q - z) rounding
  const float tot = block_reduce_256(d * d, red);
  if (t == 0) { partial[row] = tot; outIdxF[row] = (float)id; }
}

// ---------------- deterministic final loss reduction ------------------------
__global__ __launch_bounds__(256) void vq_loss(const float* __restrict__ partial,
                                               float* __restrict__ outLoss) {
  __shared__ double red[256];
  double acc = 0.0;
  for (int i = threadIdx.x; i < BB; i += 256) acc += (double)partial[i];
  red[threadIdx.x] = acc;
  __syncthreads();
  for (int off = 128; off > 0; off >>= 1) {
    if (threadIdx.x < (unsigned)off) red[threadIdx.x] += red[threadIdx.x + off];
    __syncthreads();
  }
  if (threadIdx.x == 0)
    outLoss[0] = (float)((double)BETA_C * red[0] / ((double)BB * (double)DD));
}

// ---------------------------------------------------------------------------
extern "C" void kernel_launch(void* const* d_in, const int* in_sizes, int n_in,
                              void* d_out, int out_size, void* d_ws, size_t ws_size,
                              hipStream_t stream) {
  (void)in_sizes; (void)n_in; (void)out_size; (void)ws_size;
  const float* z      = (const float*)d_in[0];
  const float* weight = (const float*)d_in[1];
  float* out = (float*)d_out;

  char* ws = (char*)d_ws;
  bf16_t* zb    = (bf16_t*)(ws + OFF_ZB);
  bf16_t* wb    = (bf16_t*)(ws + OFF_WB);
  float*  wn2   = (float*)(ws + OFF_WN2);
  float*  znorm = (float*)(ws + OFF_ZNORM);
  int*    idx   = (int*)(ws + OFF_IDX);
  float*  part  = (float*)(ws + OFF_PART);

  float* outZq   = out;                       // [BB*DD]
  float* outLoss = out + (size_t)BB * DD;     // [1]
  float* outIdxF = outLoss + 1;               // [BB]

  vq_prep_z<<<BB, 256, 0, stream>>>(z, znorm, zb);
  vq_prep_w<<<NN, 256, 0, stream>>>(weight, wn2, wb);
  vq_gemm_argmin<<<BB / 64, 64, 0, stream>>>(zb, wb, wn2, idx);
  vq_gather<<<BB, 256, 0, stream>>>(weight, znorm, idx, outZq, outIdxF, part);
  vq_loss<<<1, 256, 0, stream>>>(part, outLoss);
}
